// DFBNetconst_79688823210134
// MI455X (gfx1250) — matrix operand order repaired and checked by reading, not compiled, not run on hardware
//
#include <hip/hip_runtime.h>
#include <math.h>

// ---------------------------------------------------------------------------
// DFBNet-style unrolled proximal net on MI455X (gfx1250, wave32, WMMA).
//   u = conv(xin, W_in)                      (3 -> 64, k3, pad1)
//   repeat 15x:
//     lip2  = ||conv_t(conv(.))|| via power iteration  (fixed 32 iters)
//     gamma = 1.8 / lip2
//     tmp   = xref - conv_t(u, Wc)           (64 -> 3)
//     u     = clamp(u + gamma*conv(tmp, Wc), -l, +l)   // prox algebra fused
//   out = clip(xref - conv_t(u, W_in), 0, 1)
// Convs are implicit GEMMs on v_wmma_f32_16x16x32_bf16. Weights arrive in LDS
// via the Tensor Data Mover, then are re-laid-out ONCE into bf16 A-fragment
// order so the K-loop is pure ds_load_b128 + wmma (no per-chunk converts).
// ---------------------------------------------------------------------------

#define DEVINLINE __device__ __forceinline__

typedef __attribute__((ext_vector_type(16))) __bf16         v16bf;
typedef __attribute__((ext_vector_type(16))) unsigned short v16us;
typedef __attribute__((ext_vector_type(8)))  float          v8f;
typedef __attribute__((ext_vector_type(4)))  unsigned int   u32x4;
typedef __attribute__((ext_vector_type(8)))  int            i32x8;
typedef __attribute__((ext_vector_type(4)))  int            i32x4;

static constexpr int BB = 8, CCH = 3, HH = 256, WWD = 256, FF = 64, LLAY = 15;
static constexpr int KW    = 27;              // C*3*3 reduction depth (fwd)
static constexpr int KT    = FF * 9;          // 576 reduction depth (transpose)
static constexpr int WELEM = FF * CCH * 9;    // 1728 floats per layer's weights
static constexpr int NPOW  = 32;              // fixed power-iteration count

#if defined(__gfx1250__) && __has_builtin(__builtin_amdgcn_tensor_load_to_lds) && \
    __has_builtin(__builtin_amdgcn_s_wait_tensorcnt)
#define ATHENA_HAS_TDM 1
#endif

DEVINLINE unsigned short f2bf(float f) {            // f32 -> bf16, RNE
  unsigned u = __builtin_bit_cast(unsigned, f);
  u += 0x7FFFu + ((u >> 16) & 1u);
  return (unsigned short)(u >> 16);
}

// --- Tensor Data Mover: DMA this layer's 1728-float weight block into LDS ---
DEVINLINE void tdm_issue_weights(const float* g, void* lds) {
#ifdef ATHENA_HAS_TDM
  if (threadIdx.x < 32) {   // one wave issues; TDM ignores EXEC
    unsigned long long ga = (unsigned long long)g;
    unsigned lds_off = (unsigned)(unsigned long long)lds;   // LDS aperture low bits
    // D# group0: count=1, lds_addr, global_addr[56:0], type=2 ("image")
    u32x4 g0 = { 1u,
                 lds_off,
                 (unsigned)(ga & 0xFFFFFFFFu),
                 (unsigned)((ga >> 32) & 0x1FFFFFFu) | (2u << 30) };
    // D# group1: data_size=4B (code 2); tensor 1728 x 1; tile 1728 x 1;
    //            dim0 stride = 1728 (bit-packed per CDNA5 ISA)
    i32x8 g1 = { (int)(2u << 16),
                 (int)((WELEM & 0xFFFF) << 16),          // tensor_dim0 lo16
                 (int)(1u << 16),                        // dim0 hi=0 | tensor_dim1=1
                 (int)((WELEM & 0xFFFF) << 16),          // dim1 hi=0 | tile_dim0
                 1,                                      // tile_dim1=1
                 (int)WELEM,                             // dim0_stride lo32
                 (int)((WELEM & 0xFFFF) << 16),          // stride hi=0 | dim1_stride lo16
                 0 };
    i32x4 gz4 = { 0, 0, 0, 0 };                          // groups 2/3 unused (<=2D)
    i32x8 gz8 = { 0, 0, 0, 0, 0, 0, 0, 0 };              // extended words (unused)
    __builtin_amdgcn_tensor_load_to_lds(g0, g1, gz4, gz4, gz8, 0);
  }
#else
  (void)g; (void)lds;
#endif
}
DEVINLINE void weights_fallback_copy(const float* g, float* lds) {
#ifndef ATHENA_HAS_TDM
  for (int i = threadIdx.x; i < WELEM; i += blockDim.x) lds[i] = g[i];
#else
  (void)g; (void)lds;
#endif
}
DEVINLINE void tdm_wait() {
#ifdef ATHENA_HAS_TDM
  __builtin_amdgcn_s_wait_tensorcnt(0);  // no-op (cnt==0) on non-issuing waves
#endif
}

// ---------------------------------------------------------------------------
// conv 3 -> 64, k3 pad1.  One WMMA per wave-tile: A = 16 filters x K32(27),
// B = K32 x 16 pixels.  Block = 8 waves = 4 M-tiles x 2 N-tiles = 64ch x 32px.
// mode 0: out = conv(x)
// mode 1: out = clamp(u_in + gamma*conv(x), -l, +l)       (fused prox step)
// ---------------------------------------------------------------------------
__global__ __launch_bounds__(256)
void conv3_to_64_wmma(const float* __restrict__ x, const float* __restrict__ Wc,
                      float* __restrict__ out, const float* __restrict__ u_in,
                      const float* __restrict__ scal, const float* __restrict__ lthr,
                      int mode) {
  __shared__ float          s_w[WELEM];         // raw [64][3][3][3] f32 (TDM dest)
  __shared__ unsigned short s_wf[FF * 32];      // bf16 [64 filters][K padded to 32]
  __shared__ unsigned short s_xb[CCH * 3 * 34]; // bf16 slab: 3ch x 3rows x (32+2)

  const int b = blockIdx.z, h = blockIdx.y, w0 = blockIdx.x * 32;

  tdm_issue_weights(Wc, s_w);
  weights_fallback_copy(Wc, s_w);
  for (int i = threadIdx.x; i < CCH * 3 * 34; i += 256) {
    int c = i / 102, rem = i % 102, rr = rem / 34, col = rem % 34;
    int gh = h - 1 + rr, gw = w0 - 1 + col;
    float v = 0.f;
    if ((unsigned)gh < (unsigned)HH && (unsigned)gw < (unsigned)WWD)
      v = x[(((size_t)b * CCH + c) * HH + gh) * WWD + gw];
    s_xb[i] = f2bf(v);
  }
  tdm_wait();
  __syncthreads();                         // s_w + s_xb visible

  // one-time bf16 re-layout: s_wf[f][k] (k<27 real, 27..31 zero)
  for (int i = threadIdx.x; i < FF * 32; i += 256) {
    int f = i >> 5, k = i & 31;
    s_wf[i] = (k < KW) ? f2bf(s_w[f * KW + k]) : (unsigned short)0;
  }
  __syncthreads();

  const int lane = threadIdx.x & 31, wave = threadIdx.x >> 5;
  const int mtile = wave & 3, ntile = wave >> 2;
  const int m = lane & 15;

  // A fragment: contiguous u16 runs -> ds_load_b128-friendly, no converts
  const unsigned short* wrow = &s_wf[(mtile * 16 + m) * 32];
  const int kba = (lane < 16) ? 0 : 16;
  v16us a;
#pragma unroll
  for (int j = 0; j < 16; ++j)
    a[j] = wrow[kba + j];

  // B fragment: gathered bf16 from slab
  const int n = lane & 15;
  const int kbb = (lane < 16) ? 0 : 16;
  v16us bfr;
#pragma unroll
  for (int j = 0; j < 16; ++j) {
    int k = kbb + j;
    unsigned short v = 0;
    if (k < KW) {
      int c = k / 9, t = k % 9, rr = t / 3, ss = t % 3;
      v = s_xb[(c * 3 + rr) * 34 + ntile * 16 + n + ss];
    }
    bfr[j] = v;
  }

  v8f acc = {};
  acc = __builtin_amdgcn_wmma_f32_16x16x32_bf16(
      false, __builtin_bit_cast(v16bf, a),
      false, __builtin_bit_cast(v16bf, bfr),
      (short)0, acc, false, false);

  // D layout: VGPR r -> M=r (lanes 0-15) / M=r+8 (lanes 16-31); N = lane&15
  const int chan_base = mtile * 16 + ((lane < 16) ? 0 : 8);
  const int w = w0 + ntile * 16 + (lane & 15);
  const size_t base = (((size_t)b * FF + chan_base) * HH + h) * WWD + w;
  const size_t step = (size_t)HH * WWD;
  if (mode == 0) {
#pragma unroll
    for (int r = 0; r < 8; ++r) out[base + (size_t)r * step] = acc[r];
  } else {
    const float gamma = scal[1], l = lthr[0];
#pragma unroll
    for (int r = 0; r < 8; ++r) {
      size_t idx = base + (size_t)r * step;
      float g1 = u_in[idx] + gamma * acc[r];     // g1 - gamma*soft(g1/g, l/g)
      out[idx] = fminf(fmaxf(g1, -l), l);        //   == clamp(g1, -l, +l)
    }
  }
}

// ---------------------------------------------------------------------------
// conv_t 64 -> 3 (tied weight: I/O swapped + spatially flipped), k3 pad1.
// GEMM: M=16 (3 valid out-ch) x K=576 (18 WMMA chunks) x N=16 px per wave.
// Block = 8 waves = 128 pixels of one row; u-slab staged bf16 in LDS.
// mode 0: out = conv_t(u);  1: out = xref - conv_t(u);  2: clip(xref-.., 0, 1)
// ---------------------------------------------------------------------------
__global__ __launch_bounds__(256)
void conv64_to_3_wmma(const float* __restrict__ u, const float* __restrict__ Wc,
                      float* __restrict__ out, const float* __restrict__ xref,
                      int mode) {
  __shared__ float          s_w[WELEM];          // raw [64][3][3][3] f32 (TDM dest)
  __shared__ unsigned short s_wt[4 * KT];        // bf16 Wt[c][k], row 3 zeroed
  __shared__ unsigned short s_u[FF * 3 * 130];   // 64ch x 3rows x (128+2) bf16

  const int b = blockIdx.z, h = blockIdx.y, w0 = blockIdx.x * 128;

  tdm_issue_weights(Wc, s_w);
  weights_fallback_copy(Wc, s_w);
  for (int i = threadIdx.x; i < FF * 3 * 130; i += 256) {
    int f = i / 390, rem = i % 390, rr = rem / 130, col = rem % 130;
    int gh = h - 1 + rr, gw = w0 - 1 + col;
    float v = 0.f;
    if ((unsigned)gh < (unsigned)HH && (unsigned)gw < (unsigned)WWD) {
      const float* src = &u[(((size_t)b * FF + f) * HH + gh) * WWD + gw];
      __builtin_prefetch(src + WWD, 0, 1);        // hint next row into L2
      v = *src;
    }
    s_u[i] = f2bf(v);
  }
  tdm_wait();
  __syncthreads();                          // s_w + s_u visible

  // one-time bf16 re-layout: s_wt[c][f*9 + dh*3 + dw] = Wc[f, c, 2-dh, 2-dw]
  for (int i = threadIdx.x; i < 4 * KT; i += 256) {
    int c = i / KT, k = i - c * KT;
    unsigned short v = 0;
    if (c < CCH) {
      int f = k / 9, t = k % 9, dh = t / 3, dw = t % 3;
      v = f2bf(s_w[(f * CCH + c) * 9 + (2 - dh) * 3 + (2 - dw)]);
    }
    s_wt[i] = v;                            // row c==3 stays zero (pad lanes)
  }
  __syncthreads();

  const int lane = threadIdx.x & 31, wave = threadIdx.x >> 5;
  const int m = lane & 15;
  const int mm = (m < CCH) ? m : CCH;       // lanes m>=3 read the zero row
  const unsigned short* wtrow = &s_wt[mm * KT];
  const int kba = (lane < 16) ? 0 : 16;
  const int n = lane & 15;
  const int kbb = (lane < 16) ? 0 : 16;
  const int ncol = wave * 16 + n;

  v8f acc = {};
  for (int kc = 0; kc < 18; ++kc) {
    const int kbase = kc * 32;
    v16us a;                                // contiguous u16 runs, no converts
#pragma unroll
    for (int j = 0; j < 16; ++j)
      a[j] = wtrow[kbase + kba + j];
    v16us bfr;                              // u patches, already bf16 in LDS
#pragma unroll
    for (int j = 0; j < 16; ++j) {
      int k = kbase + kbb + j;
      int f = k / 9, t = k % 9, dh = t / 3, dw = t % 3;
      bfr[j] = s_u[(f * 3 + dh) * 130 + ncol + dw];
    }
    acc = __builtin_amdgcn_wmma_f32_16x16x32_bf16(
        false, __builtin_bit_cast(v16bf, a),
        false, __builtin_bit_cast(v16bf, bfr),
        (short)0, acc, false, false);
  }

  if (lane < 16) {                          // M rows 0..2 valid (lanes 0-15)
    const int w = w0 + wave * 16 + lane;
    const size_t base = (((size_t)b * CCH) * HH + h) * WWD + w;
    const size_t step = (size_t)HH * WWD;
    if (mode == 0) {
#pragma unroll
      for (int r = 0; r < CCH; ++r) out[base + (size_t)r * step] = acc[r];
    } else if (mode == 1) {
#pragma unroll
      for (int r = 0; r < CCH; ++r) {
        size_t idx = base + (size_t)r * step;
        out[idx] = xref[idx] - acc[r];
      }
    } else {
#pragma unroll
      for (int r = 0; r < CCH; ++r) {
        size_t idx = base + (size_t)r * step;
        float o = xref[idx] - acc[r];
        out[idx] = fminf(fmaxf(o, 0.f), 1.f);
      }
    }
  }
}

// ------------------------- small utility kernels ---------------------------
__global__ void init_pow_kernel(float* __restrict__ x, int n, unsigned seed) {
  int i = blockIdx.x * blockDim.x + threadIdx.x;
  if (i >= n) return;
  unsigned h1 = (unsigned)i * 2654435761u ^ seed;
  h1 ^= h1 >> 16; h1 *= 0x85EBCA6Bu; h1 ^= h1 >> 13; h1 *= 0xC2B2AE35u; h1 ^= h1 >> 16;
  unsigned h2 = h1 * 0x9E3779B9u + 0x7F4A7C15u;
  h2 ^= h2 >> 16; h2 *= 0x85EBCA6Bu; h2 ^= h2 >> 13;
  float u1 = ((h1 & 0xFFFFFF) + 1) * (1.0f / 16777218.0f);
  float u2 = (h2 & 0xFFFFFF) * (1.0f / 16777216.0f);
  x[i] = sqrtf(-2.0f * logf(u1)) * cosf(6.2831853f * u2);   // Box–Muller
}

__global__ void reduce_sumsq_kernel(const float* __restrict__ x, int n,
                                    float* __restrict__ parts) {
  __shared__ float sm[256];
  int tid = threadIdx.x, i = blockIdx.x * 256 + tid;
  float v = (i < n) ? x[i] * x[i] : 0.f;
  sm[tid] = v;
  __syncthreads();
  for (int s = 128; s > 0; s >>= 1) {
    if (tid < s) sm[tid] += sm[tid + s];
    __syncthreads();
  }
  if (tid == 0) parts[blockIdx.x] = sm[0];
}

__global__ void finalize_norm_kernel(const float* __restrict__ parts, int np,
                                     float* __restrict__ scal) {
  __shared__ float sm[256];
  float s = 0.f;
  for (int i = threadIdx.x; i < np; i += 256) s += parts[i];
  sm[threadIdx.x] = s;
  __syncthreads();
  for (int st = 128; st > 0; st >>= 1) {
    if (threadIdx.x < st) sm[threadIdx.x] += sm[threadIdx.x + st];
    __syncthreads();
  }
  if (threadIdx.x == 0) {
    float nv = sqrtf(sm[0]);
    scal[0] = nv;             // lip2 estimate ||conv_t(conv(x))||
    scal[1] = 1.8f / nv;      // gamma
    scal[2] = 1.0f / nv;      // normalization for next iterate
  }
}

__global__ void scale_kernel(const float* __restrict__ y, float* __restrict__ x,
                             const float* __restrict__ scal, int n) {
  int i = blockIdx.x * blockDim.x + threadIdx.x;
  if (i < n) x[i] = y[i] * scal[2];
}

// ---------------------------------------------------------------------------
extern "C" void kernel_launch(void* const* d_in, const int* in_sizes, int n_in,
                              void* d_out, int out_size, void* d_ws, size_t ws_size,
                              hipStream_t stream) {
  (void)in_sizes; (void)n_in; (void)out_size; (void)ws_size;
  const float* xref  = (const float*)d_in[0];
  const float* xin   = (const float*)d_in[1];
  const float* lthr  = (const float*)d_in[2];
  const float* W_in  = (const float*)d_in[3];
  const float* W_blk = (const float*)d_in[4];

  float* out = (float*)d_out;                       // [8,3,256,256]
  float* u   = out + (size_t)BB * CCH * HH * WWD;   // [8,64,256,256] (2nd output)

  float* tmp   = (float*)d_ws;                      // [8,3,256,256]
  float* px    = tmp + (size_t)BB * CCH * HH * WWD; // [1,3,256,256] power vec
  float* py    = px + (size_t)CCH * HH * WWD;       // [1,64,256,256]
  float* pz    = py + (size_t)FF * HH * WWD;        // [1,3,256,256]
  float* parts = pz + (size_t)CCH * HH * WWD;       // reduction partials
  float* scal  = parts + 1024;                      // {nv, gamma, 1/nv}

  const int NP = CCH * HH * WWD;                    // 196608
  const int RB = (NP + 255) / 256;                  // 768

  dim3 blk(256);
  dim3 gF(WWD / 32, HH, BB),  gF1(WWD / 32, HH, 1);
  dim3 gT(WWD / 128, HH, BB), gT1(WWD / 128, HH, 1);

  // u = in_conv(xin)
  conv3_to_64_wmma<<<gF, blk, 0, stream>>>(xin, W_in, u, nullptr, nullptr, nullptr, 0);

  for (int layer = 0; layer < LLAY; ++layer) {
    const float* Wc = W_blk + (size_t)layer * WELEM;

    // power iteration for lip2 = ||conv_t o conv|| (fixed count, deterministic)
    init_pow_kernel<<<RB, blk, 0, stream>>>(px, NP, 0x9E3779B9u * (unsigned)(layer + 1));
    reduce_sumsq_kernel<<<RB, blk, 0, stream>>>(px, NP, parts);
    finalize_norm_kernel<<<1, blk, 0, stream>>>(parts, RB, scal);
    scale_kernel<<<RB, blk, 0, stream>>>(px, px, scal, NP);
    for (int it = 0; it < NPOW; ++it) {
      conv3_to_64_wmma<<<gF1, blk, 0, stream>>>(px, Wc, py, nullptr, nullptr, nullptr, 0);
      conv64_to_3_wmma<<<gT1, blk, 0, stream>>>(py, Wc, pz, nullptr, 0);
      reduce_sumsq_kernel<<<RB, blk, 0, stream>>>(pz, NP, parts);
      finalize_norm_kernel<<<1, blk, 0, stream>>>(parts, RB, scal);
      scale_kernel<<<RB, blk, 0, stream>>>(pz, px, scal, NP);
    }

    // tmp = xref - conv_t(u);  u = clamp(u + gamma*conv(tmp), -l, +l)
    conv64_to_3_wmma<<<gT, blk, 0, stream>>>(u, Wc, tmp, xref, 1);
    conv3_to_64_wmma<<<gF, blk, 0, stream>>>(tmp, Wc, u, u, scal, lthr, 1);
  }

  // out = clip(xref - conv_t(u, W_in), 0, 1)
  conv64_to_3_wmma<<<gT, blk, 0, stream>>>(u, W_in, out, xref, 2);
}
